// Enhanced_GNN_31971736552201
// MI455X (gfx1250) — matrix operand order repaired and checked by reading, not compiled, not run on hardware
//
#include <hip/hip_runtime.h>
#include <hip/hip_bf16.h>

typedef __attribute__((ext_vector_type(16))) _Float16 v16h;
typedef __attribute__((ext_vector_type(8)))  float    v8f;

// ---------------------------------------------------------------------------
// Utility: fill float buffer
// ---------------------------------------------------------------------------
__global__ void fill_kernel(float* __restrict__ p, float v, int n) {
    int i = blockIdx.x * blockDim.x + threadIdx.x;
    if (i < n) p[i] = v;
}

// ---------------------------------------------------------------------------
// Degree / normalization precompute
// ---------------------------------------------------------------------------
__global__ void deg_kernel(const int* __restrict__ dst, float* __restrict__ deg, int E) {
    int e = blockIdx.x * blockDim.x + threadIdx.x;
    if (e < E) atomicAdd(&deg[dst[e]], 1.0f);
}

__global__ void dis_kernel(const float* __restrict__ deg, float* __restrict__ dis, int n) {
    int i = blockIdx.x * blockDim.x + threadIdx.x;
    if (i < n) dis[i] = rsqrtf(deg[i]);
}

__global__ void enorm_kernel(const int* __restrict__ src, const int* __restrict__ dst,
                             const float* __restrict__ dis, float* __restrict__ enorm, int E) {
    int e = blockIdx.x * blockDim.x + threadIdx.x;
    if (e < E) enorm[e] = dis[src[e]] * dis[dst[e]];
}

// ---------------------------------------------------------------------------
// Pre-swizzle W [K x NOUT] f32 -> f16 in the exact WMMA B-fragment register
// layout, so the GEMM loads one contiguous 32B v16h per lane per fragment.
// Fragment ft = (kb * NT + t); within a fragment, lane l (col = t*16 + l%16,
// K-half = l/16) holds elements e=0..15 with K = kb*32 + (e/8)*16 + (l/16)*8 + e%8.
// ---------------------------------------------------------------------------
__global__ void swizzle_w(const float* __restrict__ W, _Float16* __restrict__ Wsw,
                          int K, int NOUT) {
    int idx = blockIdx.x * blockDim.x + threadIdx.x;
    if (idx >= K * NOUT) return;
    const int NT = NOUT >> 4;
    int e  = idx & 15;
    int l  = (idx >> 4) & 31;
    int ft = idx >> 9;                 // fragment index
    int t  = ft % NT;
    int kb = ft / NT;
    int col = t * 16 + (l & 15);
    int kk  = kb * 32 + ((e & 8) << 1) + ((l >> 4) << 3) + (e & 7);
    Wsw[idx] = (_Float16)W[(size_t)kk * NOUT + col];
}

// ---------------------------------------------------------------------------
// WMMA GEMM: C[M x NOUT] = A[M x K] * Wsw, f32 A converted in-register,
// pre-swizzled f16 weights, f32 accumulate.
// One wave per 16-row tile; wave computes all NOUT/16 column tiles.
// Requires M%16==0, K%32==0. EXEC all-ones within active waves.
// ---------------------------------------------------------------------------
template<int NOUT>
__global__ void gemm_wmma(const float* __restrict__ A, const _Float16* __restrict__ Wsw,
                          float* __restrict__ C, int M, int K) {
    constexpr int NT = NOUT / 16;
    const int lane = threadIdx.x & 31;
    const int wave = threadIdx.x >> 5;
    const int rowTile = blockIdx.x * (blockDim.x >> 5) + wave;
    if (rowTile * 16 >= M) return;              // wave-uniform exit
    const int half = lane >> 4;                 // 0 or 1
    const int l15  = lane & 15;
    const int row  = rowTile * 16 + l15;

    const v8f zero8 = {0.f,0.f,0.f,0.f,0.f,0.f,0.f,0.f};
    v8f acc[NT];
#pragma unroll
    for (int t = 0; t < NT; ++t) acc[t] = zero8;

    for (int k0 = 0; k0 < K; k0 += 32) {
        // --- A fragment (16x32 f16 layout): lane half h holds K = {h*8..h*8+7, 16+h*8..16+h*8+7}
        const float* arow = A + (size_t)row * K + k0 + (half << 3);
        float4 a0 = *(const float4*)(arow);
        float4 a1 = *(const float4*)(arow + 4);
        float4 a2 = *(const float4*)(arow + 16);
        float4 a3 = *(const float4*)(arow + 20);
        v16h a;
        a[0]=(_Float16)a0.x;  a[1]=(_Float16)a0.y;  a[2]=(_Float16)a0.z;  a[3]=(_Float16)a0.w;
        a[4]=(_Float16)a1.x;  a[5]=(_Float16)a1.y;  a[6]=(_Float16)a1.z;  a[7]=(_Float16)a1.w;
        a[8]=(_Float16)a2.x;  a[9]=(_Float16)a2.y;  a[10]=(_Float16)a2.z; a[11]=(_Float16)a2.w;
        a[12]=(_Float16)a3.x; a[13]=(_Float16)a3.y; a[14]=(_Float16)a3.z; a[15]=(_Float16)a3.w;

        const _Float16* wbase = Wsw + ((size_t)(k0 >> 5) * NT) * 512 + (size_t)lane * 16;
#pragma unroll
        for (int t = 0; t < NT; ++t) {
            v16h b = *(const v16h*)(wbase + (size_t)t * 512);   // contiguous 32B per lane
            acc[t] = __builtin_amdgcn_wmma_f32_16x16x32_f16(
                false, a, false, b, (short)0, acc[t], false, false);
        }
    }
    // --- store: C/D layout — VGPR r, lanes 0-15 -> M=r, lanes 16-31 -> M=8+r; col=l15
#pragma unroll
    for (int t = 0; t < NT; ++t) {
#pragma unroll
        for (int r = 0; r < 8; ++r) {
            C[(size_t)(rowTile * 16 + (half << 3) + r) * NOUT + t * 16 + l15] = acc[t][r];
        }
    }
}

// ---------------------------------------------------------------------------
// z[i,c] = dis[i]^2 * hw[i,c] + bias[c]   (self-loop term + bias, scatter target)
// ---------------------------------------------------------------------------
__global__ void init_z(const float* __restrict__ hw, const float* __restrict__ dis,
                       const float* __restrict__ bias, float* __restrict__ z,
                       int Nrows, int H) {
    size_t idx = (size_t)blockIdx.x * blockDim.x + threadIdx.x;
    size_t total = (size_t)Nrows * H;
    if (idx >= total) return;
    int i = (int)(idx / H);
    int c = (int)(idx % H);
    float d = dis[i];
    z[idx] = d * d * hw[idx] + bias[c];
}

// ---------------------------------------------------------------------------
// Edge scatter-add: z[dst] += enorm[e] * hw[src]. One thread per edge.
// ---------------------------------------------------------------------------
template<int H>
__global__ void scatter_edges(const float* __restrict__ hw, const int* __restrict__ src,
                              const int* __restrict__ dst, const float* __restrict__ enorm,
                              float* __restrict__ z, int E) {
    int e = blockIdx.x * blockDim.x + threadIdx.x;
    if (e >= E) return;
    int s = src[e], d = dst[e];
    float w = enorm[e];
    const float* hs = hw + (size_t)s * H;
    float* zd = z + (size_t)d * H;
#pragma unroll
    for (int c = 0; c < H; ++c) atomicAdd(&zd[c], w * hs[c]);
}

// ---------------------------------------------------------------------------
// BatchNorm stats: sums[0..H-1]=sum, sums[H..2H-1]=sumsq  (coalesced reads,
// LDS reduce per block, then global atomics)
// ---------------------------------------------------------------------------
template<int H>
__global__ void bn_accum(const float* __restrict__ z, float* __restrict__ sums, int Nrows) {
    __shared__ float sh[2 * H];
    const int tid = threadIdx.x;
    const int perRow = blockDim.x / H;
    const int c = tid % H;
    const int rsub = tid / H;
    if (tid < 2 * H) sh[tid] = 0.0f;
    __syncthreads();
    float s = 0.f, s2 = 0.f;
    const int rowStride = gridDim.x * perRow;
    for (int r = blockIdx.x * perRow + rsub; r < Nrows; r += rowStride) {
        float v = z[(size_t)r * H + c];
        s += v; s2 += v * v;
    }
    atomicAdd(&sh[c], s);
    atomicAdd(&sh[H + c], s2);
    __syncthreads();
    if (tid < H) {
        atomicAdd(&sums[tid], sh[tid]);
        atomicAdd(&sums[H + tid], sh[H + tid]);
    }
}

// ---------------------------------------------------------------------------
// In-place BN (batch stats, biased var) + LeakyReLU(0.1)
// ---------------------------------------------------------------------------
__global__ void bn_lrelu(float* __restrict__ z, const float* __restrict__ sums,
                         const float* __restrict__ gamma, const float* __restrict__ beta,
                         int Nrows, int H) {
    size_t idx = (size_t)blockIdx.x * blockDim.x + threadIdx.x;
    size_t total = (size_t)Nrows * H;
    if (idx >= total) return;
    int c = (int)(idx % H);
    float invn = 1.0f / (float)Nrows;
    float m = sums[c] * invn;
    float var = sums[H + c] * invn - m * m;
    float val = (z[idx] - m) * rsqrtf(var + 1e-5f) * gamma[c] + beta[c];
    z[idx] = val > 0.f ? val : 0.1f * val;
}

// ---------------------------------------------------------------------------
// Pooling: per-graph count / sum / max (max via ordered-uint atomicMax)
// ---------------------------------------------------------------------------
__global__ void pool_kernel(const float* __restrict__ x3, const int* __restrict__ batch,
                            float* __restrict__ counts, float* __restrict__ psum,
                            unsigned* __restrict__ pmaxenc, int Nrows) {
    int i = blockIdx.x * blockDim.x + threadIdx.x;
    if (i >= Nrows) return;
    int g = batch[i];
    atomicAdd(&counts[g], 1.0f);
    const int H = 16;
#pragma unroll
    for (int c = 0; c < H; ++c) {
        float v = x3[(size_t)i * H + c];
        atomicAdd(&psum[g * H + c], v);
        unsigned b = __float_as_uint(v);
        unsigned enc = (b & 0x80000000u) ? ~b : (b | 0x80000000u);  // order-preserving
        atomicMax(&pmaxenc[g * H + c], enc);
    }
}

// ---------------------------------------------------------------------------
// Head: attention-weighted pool combine + MLP + sigmoid. One thread per graph.
// ---------------------------------------------------------------------------
__global__ void head_kernel(const float* __restrict__ counts, const float* __restrict__ psum,
                            const unsigned* __restrict__ pmaxenc,
                            const float* __restrict__ attn_w, const float* __restrict__ attn_b,
                            const float* __restrict__ fc1_w, const float* __restrict__ fc1_b,
                            const float* __restrict__ fc2_w, const float* __restrict__ fc2_b,
                            const float* __restrict__ out_w, const float* __restrict__ out_b,
                            float* __restrict__ out, int G) {
    int g = blockIdx.x * blockDim.x + threadIdx.x;
    if (g >= G) return;
    const int H = 16, HH = 8;
    float mean[H], mx[H], sm[H];
    float cnt = counts[g];
    float denom = cnt > 1.0f ? cnt : 1.0f;
    for (int c = 0; c < H; ++c) {
        float s = psum[g * H + c];
        unsigned enc = pmaxenc[g * H + c];
        unsigned bits = (enc & 0x80000000u) ? (enc ^ 0x80000000u) : ~enc;
        sm[c] = s;
        mean[c] = s / denom;
        mx[c] = __uint_as_float(bits);
    }
    float logit[3];
    for (int j = 0; j < 3; ++j) {
        float acc = attn_b[j];
        for (int k = 0; k < H; ++k) acc += mean[k] * attn_w[k * 3 + j];
        for (int k = 0; k < H; ++k) acc += mx[k]   * attn_w[(H + k) * 3 + j];
        for (int k = 0; k < H; ++k) acc += sm[k]   * attn_w[(2 * H + k) * 3 + j];
        logit[j] = acc;
    }
    float mmax = fmaxf(logit[0], fmaxf(logit[1], logit[2]));
    float e0 = expf(logit[0] - mmax), e1 = expf(logit[1] - mmax), e2 = expf(logit[2] - mmax);
    float inv = 1.0f / (e0 + e1 + e2);
    float a0 = e0 * inv, a1 = e1 * inv, a2 = e2 * inv;

    float xg[H], h1[H], h2[HH];
    for (int c = 0; c < H; ++c) xg[c] = a0 * mean[c] + a1 * mx[c] + a2 * sm[c];
    for (int c = 0; c < H; ++c) {
        float acc = fc1_b[c];
        for (int k = 0; k < H; ++k) acc += xg[k] * fc1_w[k * H + c];
        h1[c] = acc > 0.f ? acc : 0.1f * acc;
    }
    for (int c = 0; c < HH; ++c) {
        float acc = fc2_b[c];
        for (int k = 0; k < H; ++k) acc += h1[k] * fc2_w[k * HH + c];
        h2[c] = acc > 0.f ? acc : 0.1f * acc;
    }
    float acc = out_b[0];
    for (int k = 0; k < HH; ++k) acc += h2[k] * out_w[k];
    out[g] = 1.0f / (1.0f + expf(-acc));
}

// ---------------------------------------------------------------------------
// Launcher
// ---------------------------------------------------------------------------
static inline int cdiv(long long a, long long b) { return (int)((a + b - 1) / b); }

extern "C" void kernel_launch(void* const* d_in, const int* in_sizes, int n_in,
                              void* d_out, int out_size, void* d_ws, size_t ws_size,
                              hipStream_t stream) {
    (void)n_in; (void)ws_size;
    const float* x      = (const float*)d_in[0];
    const float* W1     = (const float*)d_in[1];
    const float* b1     = (const float*)d_in[2];
    const float* W2     = (const float*)d_in[3];
    const float* b2     = (const float*)d_in[4];
    const float* W3     = (const float*)d_in[5];
    const float* b3     = (const float*)d_in[6];
    const float* g1     = (const float*)d_in[7];
    const float* be1    = (const float*)d_in[8];
    const float* g2     = (const float*)d_in[9];
    const float* be2    = (const float*)d_in[10];
    const float* g3     = (const float*)d_in[11];
    const float* be3    = (const float*)d_in[12];
    const float* attn_w = (const float*)d_in[13];
    const float* attn_b = (const float*)d_in[14];
    const float* fc1_w  = (const float*)d_in[15];
    const float* fc1_b  = (const float*)d_in[16];
    const float* fc2_w  = (const float*)d_in[17];
    const float* fc2_b  = (const float*)d_in[18];
    const float* out_w  = (const float*)d_in[19];
    const float* out_b  = (const float*)d_in[20];
    const int*   eidx   = (const int*)d_in[21];
    const int*   batch  = (const int*)d_in[22];

    const int H1 = in_sizes[2];            // 64
    const int H2 = in_sizes[4];            // 32
    const int H3 = in_sizes[6];            // 16
    const int FIN = in_sizes[1] / H1;      // 128
    const int Nn = in_sizes[0] / FIN;      // 100000
    const int E  = in_sizes[21] / 2;       // 1600000
    const int G  = out_size;               // 64 (output is [G,1])
    const int* src = eidx;
    const int* dst = eidx + E;
    float* out = (float*)d_out;

    // -------- workspace carve (256B aligned) --------
    char* wsp = (char*)d_ws;
    auto carve = [&](size_t bytes) -> void* {
        void* p = (void*)wsp;
        wsp += (bytes + 255) & ~(size_t)255;
        return p;
    };
    float*     deg     = (float*)carve((size_t)Nn * 4);
    float*     dis     = (float*)carve((size_t)Nn * 4);
    float*     enorm   = (float*)carve((size_t)E * 4);
    float*     hw      = (float*)carve((size_t)Nn * H1 * 4);
    float*     z1      = (float*)carve((size_t)Nn * H1 * 4);
    float*     z2      = (float*)carve((size_t)Nn * H2 * 4);
    float*     z3      = (float*)carve((size_t)Nn * H3 * 4);
    float*     bnsums  = (float*)carve((size_t)2 * H1 * 4);
    float*     counts  = (float*)carve((size_t)G * 4);
    float*     psum    = (float*)carve((size_t)G * H3 * 4);
    unsigned*  pmaxenc = (unsigned*)carve((size_t)G * H3 * 4);
    _Float16*  wsw     = (_Float16*)carve((size_t)FIN * H1 * 2);  // max K*NOUT f16

    const int TB = 256;

    // -------- degree / normalization --------
    fill_kernel<<<cdiv(Nn, TB), TB, 0, stream>>>(deg, 1.0f, Nn);
    deg_kernel<<<cdiv(E, TB), TB, 0, stream>>>(dst, deg, E);
    dis_kernel<<<cdiv(Nn, TB), TB, 0, stream>>>(deg, dis, Nn);
    enorm_kernel<<<cdiv(E, TB), TB, 0, stream>>>(src, dst, dis, enorm, E);

    const int gemmBlk = 128;               // 4 waves per block
    const int tiles = Nn / 16;             // 100000/16 = 6250

    // -------- layer 1: 128 -> 64 --------
    swizzle_w<<<cdiv((long long)FIN * H1, TB), TB, 0, stream>>>(W1, wsw, FIN, H1);
    gemm_wmma<64><<<cdiv(tiles, 4), gemmBlk, 0, stream>>>(x, wsw, hw, Nn, FIN);
    init_z<<<cdiv((long long)Nn * H1, TB), TB, 0, stream>>>(hw, dis, b1, z1, Nn, H1);
    scatter_edges<64><<<cdiv(E, TB), TB, 0, stream>>>(hw, src, dst, enorm, z1, E);
    fill_kernel<<<1, 2 * 64, 0, stream>>>(bnsums, 0.0f, 2 * H1);
    bn_accum<64><<<512, TB, 0, stream>>>(z1, bnsums, Nn);
    bn_lrelu<<<cdiv((long long)Nn * H1, TB), TB, 0, stream>>>(z1, bnsums, g1, be1, Nn, H1);

    // -------- layer 2: 64 -> 32 --------
    swizzle_w<<<cdiv((long long)H1 * H2, TB), TB, 0, stream>>>(W2, wsw, H1, H2);
    gemm_wmma<32><<<cdiv(tiles, 4), gemmBlk, 0, stream>>>(z1, wsw, hw, Nn, H1);
    init_z<<<cdiv((long long)Nn * H2, TB), TB, 0, stream>>>(hw, dis, b2, z2, Nn, H2);
    scatter_edges<32><<<cdiv(E, TB), TB, 0, stream>>>(hw, src, dst, enorm, z2, E);
    fill_kernel<<<1, 2 * 64, 0, stream>>>(bnsums, 0.0f, 2 * H2);
    bn_accum<32><<<512, TB, 0, stream>>>(z2, bnsums, Nn);
    bn_lrelu<<<cdiv((long long)Nn * H2, TB), TB, 0, stream>>>(z2, bnsums, g2, be2, Nn, H2);

    // -------- layer 3: 32 -> 16 --------
    swizzle_w<<<cdiv((long long)H2 * H3, TB), TB, 0, stream>>>(W3, wsw, H2, H3);
    gemm_wmma<16><<<cdiv(tiles, 4), gemmBlk, 0, stream>>>(z2, wsw, hw, Nn, H2);
    init_z<<<cdiv((long long)Nn * H3, TB), TB, 0, stream>>>(hw, dis, b3, z3, Nn, H3);
    scatter_edges<16><<<cdiv(E, TB), TB, 0, stream>>>(hw, src, dst, enorm, z3, E);
    fill_kernel<<<1, 2 * 64, 0, stream>>>(bnsums, 0.0f, 2 * H3);
    bn_accum<16><<<512, TB, 0, stream>>>(z3, bnsums, Nn);
    bn_lrelu<<<cdiv((long long)Nn * H3, TB), TB, 0, stream>>>(z3, bnsums, g3, be3, Nn, H3);

    // -------- pooling + head --------
    fill_kernel<<<1, TB, 0, stream>>>(counts, 0.0f, G);
    fill_kernel<<<cdiv((long long)G * H3, TB), TB, 0, stream>>>(psum, 0.0f, G * H3);
    fill_kernel<<<cdiv((long long)G * H3, TB), TB, 0, stream>>>((float*)pmaxenc, 0.0f, G * H3);
    pool_kernel<<<cdiv(Nn, TB), TB, 0, stream>>>(z3, batch, counts, psum, pmaxenc, Nn);
    head_kernel<<<cdiv(G, 64), 64, 0, stream>>>(counts, psum, pmaxenc,
                                                attn_w, attn_b, fc1_w, fc1_b,
                                                fc2_w, fc2_b, out_w, out_b, out, G);
}
